// PINN_Train_78735340470335
// MI455X (gfx1250) — matrix operand order repaired and checked by reading, not compiled, not run on hardware
//
#include <hip/hip_runtime.h>

// PINN loss on MI455X: all hidden-layer GEMMs via v_wmma_f32_16x16x32_f16
// (f32 accumulate), weights pre-transposed to f16 in d_ws so every B fragment
// is two contiguous global b128 loads; activations staged in LDS so every A
// fragment is two ds_load_b128.

#define LS 264  // 256 + 8 halves of pad -> rows stay 16B aligned, avoids bank conflicts

typedef _Float16 h16;
typedef __attribute__((ext_vector_type(8)))  _Float16 v8h;
typedef __attribute__((ext_vector_type(16))) _Float16 v16h;
typedef __attribute__((ext_vector_type(8)))  float    v8f;

__device__ __forceinline__ v16h ld_frag(const h16* p, int hiOff) {
  v8h lo = *(const v8h*)(p);
  v8h hi = *(const v8h*)(p + hiOff);
  return __builtin_shufflevector(lo, hi, 0,1,2,3,4,5,6,7,8,9,10,11,12,13,14,15);
}

__device__ __forceinline__ v8f wmma16(v16h a, v16h b, v8f c) {
  return __builtin_amdgcn_wmma_f32_16x16x32_f16(false, a, false, b, (short)0, c,
                                                false, false);
}

// ---------------- prep kernels ----------------
__global__ void k_zero3(float* o) { if (threadIdx.x < 3) o[threadIdx.x] = 0.0f; }

// WT[n][k] = (f16) W[k][n] : B-fragment lane (=column n) reads contiguous K.
__global__ void k_transpose_w(const float* __restrict__ W, h16* __restrict__ WT) {
  int n = blockIdx.x, k = threadIdx.x;
  WT[n * 256 + k] = (h16)W[k * 256 + n];
}

__global__ void k_copy_b(const float* __restrict__ b, float* __restrict__ dst) {
  dst[threadIdx.x] = b[threadIdx.x];
}

// ---------------- forward (bc / data) kernel ----------------
// One block = 16 points. 8 waves, each wave owns two 16-column tiles.
__global__ __launch_bounds__(256) void fwd_kernel(
    const float* __restrict__ X, int n,
    const float* __restrict__ W0, const float* __restrict__ b0,
    const h16*  __restrict__ WT, const float* __restrict__ bpack,
    const float* __restrict__ W6, const float* __restrict__ b6,
    const float* __restrict__ realp,
    float t0, float t1, float t2, float t3,
    float s0, float s1, float s2, float s3,
    float* __restrict__ out_slot) {
  __shared__ h16   H[16 * LS];
  __shared__ float Xs[16][2];
  __shared__ float red[64];

  const int tid  = threadIdx.x;
  const int lane = tid & 31;
  const int wv   = tid >> 5;
  const int idx0 = blockIdx.x * 16;

  if (tid < 32) {
    int m = tid >> 1, c = tid & 1, gi = idx0 + m;
    Xs[m][c] = (gi < n) ? X[gi * 2 + c] : 0.0f;
  }
  __syncthreads();

  // layer 0: 2 -> 256 (cheap, VALU)
  for (int e = tid; e < 4096; e += 256) {
    int m = e >> 8, nn = e & 255;
    float a = Xs[m][0] * W0[nn] + Xs[m][1] * W0[256 + nn] + b0[nn];
    H[m * LS + nn] = (h16)tanhf(a);
  }
  __syncthreads();

  const int row  = lane & 15;
  const int kh   = (lane >> 4) * 16;   // A: K half select
  const int colL = lane & 15;
  const int kb0  = (lane >> 4) * 16;   // B: K half select
  const int mb   = (lane >> 4) * 8;    // D: row base

  for (int l = 0; l < 5; ++l) {
    const h16*   Wl = WT + l * 65536;
    const float* bl = bpack + l * 256;
    if (lane == 0 && l < 4) __builtin_prefetch(WT + (l + 1) * 65536, 0, 0);

    v8f acc0 = {}, acc1 = {};
    const int nb0 = (2 * wv) * 16, nb1 = nb0 + 16;
    for (int kc = 0; kc < 8; ++kc) {
      int ko = kc * 32;
      v16h a  = ld_frag(&H[row * LS + ko + kh], 8);
      v16h bA = ld_frag(&Wl[(nb0 + colL) * 256 + ko + kb0], 8);
      v16h bB = ld_frag(&Wl[(nb1 + colL) * 256 + ko + kb0], 8);
      acc0 = wmma16(a, bA, acc0);
      acc1 = wmma16(a, bB, acc1);
    }
    __syncthreads();  // all waves done reading H -> safe to overwrite
    float bv0 = bl[nb0 + colL], bv1 = bl[nb1 + colL];
#pragma unroll
    for (int v = 0; v < 8; ++v) {
      H[(mb + v) * LS + nb0 + colL] = (h16)tanhf(acc0[v] + bv0);
      H[(mb + v) * LS + nb1 + colL] = (h16)tanhf(acc1[v] + bv1);
    }
    __syncthreads();
  }

  // output layer 256 -> 4 + per-column MSE contribution
  float contrib = 0.0f;
  if (tid < 64) {
    int m = tid >> 2, j = tid & 3;
    float o = b6[j];
    for (int k = 0; k < 256; ++k) o += (float)H[m * LS + k] * W6[k * 4 + j];
    int gi = idx0 + m;
    if (gi < n) {
      float tgt, scl;
      if (realp) { tgt = realp[gi * 4 + j]; scl = s0; }
      else {
        tgt = (j == 0) ? t0 : (j == 1) ? t1 : (j == 2) ? t2 : t3;
        scl = (j == 0) ? s0 : (j == 1) ? s1 : (j == 2) ? s2 : s3;
      }
      float d = o - tgt;
      contrib = scl * d * d;
    }
    red[tid] = contrib;
  }
  __syncthreads();
  if (tid == 0) {
    float s = 0.0f;
    for (int i = 0; i < 64; ++i) s += red[i];
    atomicAdd(out_slot, s);
  }
}

// ---------------- PDE kernel: primal + 4 tangent streams ----------------
__global__ __launch_bounds__(256) void pde_kernel(
    const float* __restrict__ X, int n,
    const float* __restrict__ W0, const float* __restrict__ b0,
    const h16*  __restrict__ WT, const float* __restrict__ bpack,
    const float* __restrict__ W6, const float* __restrict__ b6,
    float scale, float* __restrict__ out_slot) {
  __shared__ h16   Hs[5][16 * LS];  // primal, hz, hr, hzz, hrr
  __shared__ float Xs[16][2];
  __shared__ float Os[5][16][4];
  __shared__ float red[16];

  const int tid  = threadIdx.x;
  const int lane = tid & 31;
  const int wv   = tid >> 5;
  const int idx0 = blockIdx.x * 16;

  if (tid < 32) {
    int m = tid >> 1, c = tid & 1, gi = idx0 + m;
    Xs[m][c] = (gi < n) ? X[gi * 2 + c] : 0.0f;
  }
  __syncthreads();

  // layer 0: a = x@W0 + b0 ; a_z = W0[0,:], a_r = W0[1,:], a_zz = a_rr = 0
  for (int e = tid; e < 4096; e += 256) {
    int m = e >> 8, nn = e & 255;
    float wz = W0[nn], wr = W0[256 + nn];
    float a  = Xs[m][0] * wz + Xs[m][1] * wr + b0[nn];
    float t  = tanhf(a);
    float sd = 1.0f - t * t;           // tanh'
    int off = m * LS + nn;
    Hs[0][off] = (h16)t;
    Hs[1][off] = (h16)(sd * wz);
    Hs[2][off] = (h16)(sd * wr);
    Hs[3][off] = (h16)(-2.0f * t * sd * wz * wz);  // tanh''*(a')^2
    Hs[4][off] = (h16)(-2.0f * t * sd * wr * wr);
  }
  __syncthreads();

  const int row  = lane & 15;
  const int kh   = (lane >> 4) * 16;
  const int colL = lane & 15;
  const int kb0  = (lane >> 4) * 16;
  const int mb   = (lane >> 4) * 8;

  for (int l = 0; l < 5; ++l) {
    const h16*   Wl = WT + l * 65536;
    const float* bl = bpack + l * 256;
    if (lane == 0 && l < 4) __builtin_prefetch(WT + (l + 1) * 65536, 0, 0);

    v8f acc[2][5];
#pragma unroll
    for (int ti = 0; ti < 2; ++ti)
#pragma unroll
      for (int s = 0; s < 5; ++s) acc[ti][s] = (v8f){};

    for (int kc = 0; kc < 8; ++kc) {
      int ko = kc * 32;
      v16h af[5];
#pragma unroll
      for (int s = 0; s < 5; ++s)
        af[s] = ld_frag(&Hs[s][row * LS + ko + kh], 8);
#pragma unroll
      for (int ti = 0; ti < 2; ++ti) {
        int col = (2 * wv + ti) * 16 + colL;
        v16h bf = ld_frag(&Wl[col * 256 + ko + kb0], 8);  // B reused across 5 streams
#pragma unroll
        for (int s = 0; s < 5; ++s) acc[ti][s] = wmma16(af[s], bf, acc[ti][s]);
      }
    }
    __syncthreads();  // all reads of Hs complete

#pragma unroll
    for (int ti = 0; ti < 2; ++ti) {
      int col = (2 * wv + ti) * 16 + colL;
      float bv = bl[col];
#pragma unroll
      for (int v = 0; v < 8; ++v) {
        float t   = tanhf(acc[ti][0][v] + bv);
        float sd  = 1.0f - t * t;
        float az  = acc[ti][1][v], ar  = acc[ti][2][v];
        float azz = acc[ti][3][v], arr = acc[ti][4][v];
        int off = (mb + v) * LS + col;
        Hs[0][off] = (h16)t;
        Hs[1][off] = (h16)(sd * az);
        Hs[2][off] = (h16)(sd * ar);
        Hs[3][off] = (h16)(sd * azz - 2.0f * t * sd * az * az);
        Hs[4][off] = (h16)(sd * arr - 2.0f * t * sd * ar * ar);
      }
    }
    __syncthreads();
  }

  // final linear layer 256 -> 4 for all five streams
  if (tid < 64) {
    int m = tid >> 2, j = tid & 3;
    float o0 = b6[j], o1 = 0.0f, o2 = 0.0f, o3 = 0.0f, o4 = 0.0f;
    for (int k = 0; k < 256; ++k) {
      float w = W6[k * 4 + j];
      int off = m * LS + k;
      o0 += (float)Hs[0][off] * w;
      o1 += (float)Hs[1][off] * w;
      o2 += (float)Hs[2][off] * w;
      o3 += (float)Hs[3][off] * w;
      o4 += (float)Hs[4][off] * w;
    }
    Os[0][m][j] = o0; Os[1][m][j] = o1; Os[2][m][j] = o2;
    Os[3][m][j] = o3; Os[4][m][j] = o4;
  }
  __syncthreads();

  if (tid < 16) {
    int m = tid, gi = idx0 + m;
    float c = 0.0f;
    if (gi < n) {
      const float NU = 8.0e-4f;  // MIU/(U*L) = 1e-6/(0.025*0.05)
      float r = Xs[m][1], ir = 1.0f / r;
      float uz = Os[0][m][0], us = Os[0][m][1], ur = Os[0][m][2];
      float uz_z = Os[1][m][0], us_z = Os[1][m][1], ur_z = Os[1][m][2], p_z = Os[1][m][3];
      float uz_r = Os[2][m][0], us_r = Os[2][m][1], ur_r = Os[2][m][2], p_r = Os[2][m][3];
      float uz_zz = Os[3][m][0], us_zz = Os[3][m][1], ur_zz = Os[3][m][2];
      float uz_rr = Os[4][m][0], us_rr = Os[4][m][1], ur_rr = Os[4][m][2];
      float eq1 = ur*ur_r + uz*ur_z - us*us*ir + p_r - NU*ir*ur_r - NU*ur_rr - NU*ur_zz + NU*ur*ir*ir;
      float eq2 = ur*us_r + uz*us_z + ur*us*ir - NU*ir*us_r - NU*us_rr - NU*us_zz + NU*us*ir*ir;
      float eq3 = ur*uz_r + uz*uz_z + p_z - NU*ir*uz_r - NU*uz_rr - NU*uz_zz;
      float eq4 = ur + r*ur_r + r*uz_z;
      c = (eq1*eq1 + eq2*eq2 + eq3*eq3 + eq4*eq4) * scale;
    }
    red[tid] = c;
  }
  __syncthreads();
  if (tid == 0) {
    float s = 0.0f;
    for (int i = 0; i < 16; ++i) s += red[i];
    atomicAdd(out_slot, s);
  }
}

// ---------------- host ----------------
extern "C" void kernel_launch(void* const* d_in, const int* in_sizes, int n_in,
                              void* d_out, int out_size, void* d_ws, size_t ws_size,
                              hipStream_t stream) {
  const float* W[7]; const float* B[7];
  for (int i = 0; i < 7; ++i) {
    W[i] = (const float*)d_in[2 * i];
    B[i] = (const float*)d_in[2 * i + 1];
  }
  const float* bc1  = (const float*)d_in[14];
  const float* bc2  = (const float*)d_in[15];
  const float* bc3  = (const float*)d_in[16];
  const float* bc4  = (const float*)d_in[17];
  const float* bc5  = (const float*)d_in[18];
  const float* f1   = (const float*)d_in[19];
  const float* xin  = (const float*)d_in[20];
  const float* real = (const float*)d_in[21];
  float* out = (float*)d_out;

  h16*   WTp   = (h16*)d_ws;
  float* bpack = (float*)((char*)d_ws + (size_t)5 * 65536 * sizeof(h16));

  k_zero3<<<1, 32, 0, stream>>>(out);
  for (int l = 0; l < 5; ++l) {
    k_transpose_w<<<256, 256, 0, stream>>>(W[l + 1], WTp + (size_t)l * 65536);
    k_copy_b<<<1, 256, 0, stream>>>(B[l + 1], bpack + l * 256);
  }

  int n1 = in_sizes[14] / 2, n2 = in_sizes[15] / 2, n3 = in_sizes[16] / 2;
  int n4 = in_sizes[17] / 2, n5 = in_sizes[18] / 2;
  int nf = in_sizes[19] / 2, nd = in_sizes[20] / 2;
  float i1 = 1.0f / n1, i2 = 1.0f / n2, i3 = 1.0f / n3, i4 = 1.0f / n4, i5 = 1.0f / n5;

#define GRID(N) dim3(((N) + 15) / 16)
  fwd_kernel<<<GRID(n1), 256, 0, stream>>>(bc1, n1, W[0], B[0], WTp, bpack, W[6], B[6],
      nullptr, 0.f, 0.f, 0.f, 0.f, i1, i1, i1, 0.f, out + 0);
  fwd_kernel<<<GRID(n2), 256, 0, stream>>>(bc2, n2, W[0], B[0], WTp, bpack, W[6], B[6],
      nullptr, 0.f, 0.f, 0.f, 0.f, i2, i2, i2, 0.f, out + 0);
  fwd_kernel<<<GRID(n3), 256, 0, stream>>>(bc3, n3, W[0], B[0], WTp, bpack, W[6], B[6],
      nullptr, 0.f, 1.f, 0.f, 0.f, i3, i3, i3, 0.f, out + 0);
  fwd_kernel<<<GRID(n4), 256, 0, stream>>>(bc4, n4, W[0], B[0], WTp, bpack, W[6], B[6],
      nullptr, 0.f, 0.f, 0.f, 0.f, i4, i4, i4, 0.f, out + 0);
  fwd_kernel<<<GRID(n5), 256, 0, stream>>>(bc5, n5, W[0], B[0], WTp, bpack, W[6], B[6],
      nullptr, 0.f, 0.f, 0.f, 0.f, 0.f, 0.f, 0.f, i5, out + 0);
  fwd_kernel<<<GRID(nd), 256, 0, stream>>>(xin, nd, W[0], B[0], WTp, bpack, W[6], B[6],
      real, 0.f, 0.f, 0.f, 0.f, 0.25f / nd, 0.f, 0.f, 0.f, out + 1);
  pde_kernel<<<GRID(nf), 256, 0, stream>>>(f1, nf, W[0], B[0], WTp, bpack, W[6], B[6],
      0.25f / nf, out + 2);
#undef GRID
}
